// Net_58506044506146
// MI455X (gfx1250) — matrix operand order repaired and checked by reading, not compiled, not run on hardware
//
#include <hip/hip_runtime.h>
#include <hip/hip_bf16.h>
#include <stdint.h>

#define NU_ 30000
#define NH_ 10000
#define NV_ 60000
#define NN_ (NU_ + NH_ + NV_)   /* 100000 */
#define D_  128
#define DF_ 512
#define B_  4096

typedef __attribute__((ext_vector_type(16))) __bf16 v16bf;
typedef __attribute__((ext_vector_type(8)))  float  v8f;

__device__ __forceinline__ unsigned short f2bf(float f) {
  unsigned u = __float_as_uint(f);
  u += 0x7FFFu + ((u >> 16) & 1u);          // round-to-nearest-even
  return (unsigned short)(u >> 16);
}
__device__ __forceinline__ float leakyf(float v, float s) { return v >= 0.f ? v : s * v; }
// order-preserving float<->uint encoding for atomicMax on floats
__device__ __forceinline__ unsigned encf(float f) {
  unsigned b = __float_as_uint(f);
  return (b & 0x80000000u) ? ~b : (b | 0x80000000u);
}
__device__ __forceinline__ float decf(unsigned u) {
  return (u & 0x80000000u) ? __uint_as_float(u ^ 0x80000000u) : __uint_as_float(~u);
}

// ---------------- elementwise utility kernels ----------------
__global__ void fill_f32(float* __restrict__ p, float v, int n) {
  int t = blockIdx.x * blockDim.x + threadIdx.x;
  if (t < n) p[t] = v;
}
__global__ void fill_u32(unsigned* __restrict__ p, unsigned v, int n) {
  int t = blockIdx.x * blockDim.x + threadIdx.x;
  if (t < n) p[t] = v;
}
__global__ void copy_f32(float* __restrict__ dst, const float* __restrict__ src, int n) {
  int t = blockIdx.x * blockDim.x + threadIdx.x;
  if (t < n) dst[t] = src[t];
}
// bf16 convert, no transpose (W_trans is already [n][k] for our B layout)
__global__ void conv_bf16(const float* __restrict__ src, unsigned short* __restrict__ dst, int n) {
  int t = blockIdx.x * blockDim.x + threadIdx.x;
  if (t < n) dst[t] = f2bf(src[t]);
}
// bf16 convert with transpose for 128x128 weights: dst[n*128+k] = W[k*128+n]
__global__ void convT128_bf16(const float* __restrict__ src, unsigned short* __restrict__ dst) {
  int t = blockIdx.x * blockDim.x + threadIdx.x;
  if (t >= 128 * 128) return;
  int n = t >> 7, k = t & 127;
  dst[t] = f2bf(src[k * 128 + n]);
}

// ---------------- WMMA GEMM: Out[rows][128] = act( A[rows][K] @ Bt + Cadd + bias ) ----------------
// Bt is bf16, layout [n][k] (i.e. transposed weight), K in {128, 512}.
// 32 output rows per block; each wave owns a 16-col tile with two M-strips sharing one
// B fragment (2 WMMA per B load), software-pipelined with two fragment sets.
#define LOAD_FRAGS(k0_, fa0_, fa1_, fb_)                                   \
  do {                                                                     \
    int ka_ = (k0_) + hi * 8;                                              \
    _Pragma("unroll")                                                      \
    for (int i_ = 0; i_ < 4; ++i_) {                                       \
      fa0_.u[i_]     = *(const unsigned*)&ap0[ka_ + 2 * i_];               \
      fa0_.u[4 + i_] = *(const unsigned*)&ap0[ka_ + 16 + 2 * i_];          \
      fa1_.u[i_]     = *(const unsigned*)&ap1[ka_ + 2 * i_];               \
      fa1_.u[4 + i_] = *(const unsigned*)&ap1[ka_ + 16 + 2 * i_];          \
    }                                                                      \
    _Pragma("unroll")                                                      \
    for (int j_ = 0; j_ < 8; ++j_)                                         \
      fb_.u[j_] = *(const unsigned*)&bp[ka_ + (j_ >> 2) * 16 + 2 * (j_ & 3)]; \
  } while (0)

__global__ __launch_bounds__(256)
void gemm_bf16_wmma(const float* __restrict__ A, const unsigned short* __restrict__ Bt,
                    const float* __restrict__ Cadd, const float* __restrict__ bias,
                    float* __restrict__ Out, int rows, int K, int kshift, int act) {
  __shared__ unsigned short As[32 * 512];          // 32KB, covers K up to 512
  const int row0 = blockIdx.x * 32;
  const int tid  = threadIdx.x;

  // cooperatively stage + convert 32xK A tile into LDS (bf16), clamped at rows
  const int total = 32 << kshift;
  for (int i = tid; i < total; i += 256) {
    int r = i >> kshift, k = i & (K - 1);
    int rr = row0 + r; if (rr >= rows) rr = rows - 1;
    As[i] = f2bf(A[(size_t)rr * K + k]);
  }
  __syncthreads();

  const int wave = tid >> 5, lane = tid & 31;
  const int cn   = wave * 16;                      // this wave's 16-column tile
  const int mrow = lane & 15;
  const int hi   = lane >> 4;
  const bool has2 = (row0 + 16) < rows;

  // A fragment: 16x32 bf16, lanes 0-15 rows M=lane (K 0..7,16..23), lanes 16-31 K 8..15,24..31
  const unsigned short* ap0 = &As[mrow * K];
  const unsigned short* ap1 = &As[(16 + mrow) * K];
  // B fragment: 32x16 bf16, lane holds column n = cn + (lane&15); lanes 0-15 K 0..7,16..23, 16-31 K 8..15,24..31
  const unsigned short* bp  = &Bt[(size_t)(cn + mrow) * K];

  union Frag { v16bf v; unsigned u[8]; };
  Frag a0A, a1A, bA, a0B, a1B, bB;

  v8f c0 = {}, c1 = {};
  LOAD_FRAGS(0, a0A, a1A, bA);
  for (int k0 = 0; k0 < K; k0 += 64) {             // two 32-K steps per iteration
    LOAD_FRAGS(k0 + 32, a0B, a1B, bB);
    c0 = __builtin_amdgcn_wmma_f32_16x16x32_bf16(false, a0A.v, false, bA.v, (short)0, c0, false, false);
    c1 = __builtin_amdgcn_wmma_f32_16x16x32_bf16(false, a1A.v, false, bA.v, (short)0, c1, false, false);
    if (k0 + 64 < K) LOAD_FRAGS(k0 + 64, a0A, a1A, bA);
    c0 = __builtin_amdgcn_wmma_f32_16x16x32_bf16(false, a0B.v, false, bB.v, (short)0, c0, false, false);
    c1 = __builtin_amdgcn_wmma_f32_16x16x32_bf16(false, a1B.v, false, bB.v, (short)0, c1, false, false);
  }

  // C/D layout: lane 0-15 -> N=lane, M=r; lane 16-31 -> N=lane-16, M=r+8
  const int coln = cn + (lane & 15);
#pragma unroll
  for (int r = 0; r < 8; ++r) {
    int m   = r + hi * 8;
    int idx = (row0 + m) * D_ + coln;
    float v = c0[r];
    if (Cadd) v += Cadd[idx];
    if (bias) v += bias[coln];
    if (act)  v = leakyf(v, 0.01f);
    Out[idx] = v;
  }
  if (has2) {
#pragma unroll
    for (int r = 0; r < 8; ++r) {
      int m   = r + hi * 8;
      int idx = (row0 + 16 + m) * D_ + coln;
      float v = c1[r];
      if (Cadd) v += Cadd[idx];
      if (bias) v += bias[coln];
      if (act)  v = leakyf(v, 0.01f);
      Out[idx] = v;
    }
  }
}

// ---------------- row-wise kernels (wave per row, 128 cols) ----------------
__global__ __launch_bounds__(256)
void l2norm_rows(float* __restrict__ x, int n) {
  int wid  = (blockIdx.x * blockDim.x + threadIdx.x) >> 5;
  int lane = threadIdx.x & 31;
  if (wid >= n) return;
  float* r = x + (size_t)wid * D_;
  float s = 0.f;
#pragma unroll
  for (int j = 0; j < 4; ++j) { float v = r[lane + 32 * j]; s += v * v; }
  for (int o = 16; o; o >>= 1) s += __shfl_xor(s, o, 32);
  float inv = 1.f / fmaxf(sqrtf(s), 1e-12f);
#pragma unroll
  for (int j = 0; j < 4; ++j) r[lane + 32 * j] *= inv;
}

__global__ __launch_bounds__(256)
void gat_node_att(const float* __restrict__ h, const float* __restrict__ att,
                  float* __restrict__ a_dst, float* __restrict__ a_src, int n) {
  int wid  = (blockIdx.x * blockDim.x + threadIdx.x) >> 5;
  int lane = threadIdx.x & 31;
  if (wid >= n) return;
  const float* hr = h + (size_t)wid * D_;
  float s0 = 0.f, s1 = 0.f;
#pragma unroll
  for (int j = 0; j < 4; ++j) {
    int c = lane + 32 * j;
    float v = hr[c];
    s0 += v * att[c];
    s1 += v * att[D_ + c];
  }
  for (int o = 16; o; o >>= 1) { s0 += __shfl_xor(s0, o, 32); s1 += __shfl_xor(s1, o, 32); }
  if (lane == 0) { a_dst[wid] = s0; a_src[wid] = s1; }
}

// ---------------- GAT edge-scatter kernels (edges + implicit self loops) ----------------
__global__ void gat_alpha_max(const int* __restrict__ src, const int* __restrict__ dst, int E, int n,
                              const float* __restrict__ a_dst, const float* __restrict__ a_src,
                              unsigned* __restrict__ m_u) {
  int t = blockIdx.x * blockDim.x + threadIdx.x;
  if (t >= E + n) return;
  int s, d;
  if (t < E) { s = src[t]; d = dst[t]; } else { s = d = t - E; }
  float a = leakyf(a_dst[d] + a_src[s], 0.2f);
  atomicMax(&m_u[d], encf(a));
}

__global__ void gat_exp_sum(const int* __restrict__ src, const int* __restrict__ dst, int E, int n,
                            const float* __restrict__ a_dst, const float* __restrict__ a_src,
                            const unsigned* __restrict__ m_u, float* __restrict__ ssum) {
  int t = blockIdx.x * blockDim.x + threadIdx.x;
  if (t >= E + n) return;
  int s, d;
  if (t < E) { s = src[t]; d = dst[t]; } else { s = d = t - E; }
  float a = leakyf(a_dst[d] + a_src[s], 0.2f);
  atomicAdd(&ssum[d], __expf(a - decf(m_u[d])));
}

__global__ __launch_bounds__(256)
void gat_aggregate(const int* __restrict__ src, const int* __restrict__ dst, int E, int n,
                   const float* __restrict__ h, const float* __restrict__ a_dst,
                   const float* __restrict__ a_src, const unsigned* __restrict__ m_u,
                   const float* __restrict__ ssum, float* __restrict__ agg) {
  int wid  = (blockIdx.x * blockDim.x + threadIdx.x) >> 5;
  int lane = threadIdx.x & 31;
  if (wid >= E + n) return;
  int s, d;
  if (wid < E) { s = src[wid]; d = dst[wid]; } else { s = d = wid - E; }
  const float* hs = h + (size_t)s * D_;
  __builtin_prefetch(hs, 0, 1);                     // global_prefetch_b8
  float w;
  if (lane == 0) {
    float a = leakyf(a_dst[d] + a_src[s], 0.2f);
    float e = __expf(a - decf(m_u[d]));
    w = e / (ssum[d] + 1e-16f);
  }
  w = __shfl(w, 0, 32);
  float* ad = agg + (size_t)d * D_;
#pragma unroll
  for (int j = 0; j < 4; ++j) {
    int c = lane + 32 * j;
    atomicAdd(&ad[c], hs[c] * w);
  }
}

__global__ void gat_finalize(const float* __restrict__ agg, const float* __restrict__ b,
                             float* __restrict__ x, int n) {
  int t = blockIdx.x * blockDim.x + threadIdx.x;
  if (t >= n) return;
  x[t] = leakyf(agg[t] + b[t & 127], 0.01f);
}

// ---------------- SAGE kernels ----------------
__global__ __launch_bounds__(256)
void sage_aggregate(const int* __restrict__ src, const int* __restrict__ dst, int E, int n,
                    const float* __restrict__ x, float* __restrict__ agg, float* __restrict__ deg) {
  int wid  = (blockIdx.x * blockDim.x + threadIdx.x) >> 5;
  int lane = threadIdx.x & 31;
  if (wid >= E + n) return;
  int s, d;
  if (wid < E) { s = src[wid]; d = dst[wid]; } else { s = d = wid - E; }
  const float* xs = x + (size_t)s * D_;
  __builtin_prefetch(xs, 0, 1);
  if (lane == 0) atomicAdd(&deg[d], 1.f);
  float* ad = agg + (size_t)d * D_;
#pragma unroll
  for (int j = 0; j < 4; ++j) {
    int c = lane + 32 * j;
    atomicAdd(&ad[c], xs[c]);
  }
}

__global__ void sage_norm(float* __restrict__ agg, const float* __restrict__ deg, int n) {
  int t = blockIdx.x * blockDim.x + threadIdx.x;
  if (t >= n) return;
  agg[t] /= fmaxf(deg[t >> 7], 1.f);
}

// ---------------- final stage kernels ----------------
__global__ void gather_item_rows(const float* __restrict__ x, const int* __restrict__ item,
                                 int col, float* __restrict__ out) {
  int t = blockIdx.x * blockDim.x + threadIdx.x;
  if (t >= B_ * D_) return;
  int b = t >> 7, c = t & 127;
  int node = item[b * 4 + col];
  out[t] = x[(size_t)node * D_ + c];
}
__global__ void gather_vid(const float* __restrict__ vf, const int* __restrict__ item,
                           float* __restrict__ out) {
  int t = blockIdx.x * blockDim.x + threadIdx.x;
  if (t >= B_ * DF_) return;
  int b = t >> 9, c = t & 511;
  int node = item[b * 4 + 1] - (NU_ + NH_);
  out[t] = vf[(size_t)node * DF_ + c];
}
__global__ __launch_bounds__(256)
void scores_kernel(const float* __restrict__ usv, const float* __restrict__ usp,
                   const float* __restrict__ usn, float* __restrict__ out) {
  int wid  = (blockIdx.x * blockDim.x + threadIdx.x) >> 5;
  int lane = threadIdx.x & 31;
  if (wid >= B_) return;
  const float* v = usv + (size_t)wid * D_;
  const float* p = usp + (size_t)wid * D_;
  const float* q = usn + (size_t)wid * D_;
  float sp = 0.f, sn = 0.f;
#pragma unroll
  for (int j = 0; j < 4; ++j) {
    int c = lane + 32 * j;
    float a = v[c];
    sp += a * p[c];
    sn += a * q[c];
  }
  for (int o = 16; o; o >>= 1) { sp += __shfl_xor(sp, o, 32); sn += __shfl_xor(sn, o, 32); }
  if (lane == 0) { out[wid] = sp; out[B_ + wid] = sn; }
}

// ---------------- host driver ----------------
extern "C" void kernel_launch(void* const* d_in, const int* in_sizes, int n_in,
                              void* d_out, int out_size, void* d_ws, size_t ws_size,
                              hipStream_t stream) {
  (void)n_in; (void)out_size; (void)ws_size;
  const int*   item    = (const int*)d_in[0];
  const int*   uh_ei   = (const int*)d_in[1];
  const int*   v_ei    = (const int*)d_in[2];
  const float* vf      = (const float*)d_in[3];
  const float* uh_emb  = (const float*)d_in[4];
  const float* W_trans = (const float*)d_in[5];
  const float* b_trans = (const float*)d_in[6];
  const float* gatW[3]   = {(const float*)d_in[7],  (const float*)d_in[12], (const float*)d_in[17]};
  const float* gatAtt[3] = {(const float*)d_in[8],  (const float*)d_in[13], (const float*)d_in[18]};
  const float* gatB[3]   = {(const float*)d_in[9],  (const float*)d_in[14], (const float*)d_in[19]};
  const float* sageW[3]  = {(const float*)d_in[10], (const float*)d_in[15], (const float*)d_in[20]};
  const float* sageB[3]  = {(const float*)d_in[11], (const float*)d_in[16], (const float*)d_in[21]};
  const float* weight_v   = (const float*)d_in[22];
  const float* weight_h   = (const float*)d_in[23];
  const float* weight_v_u = (const float*)d_in[24];
  const float* weight_h_u = (const float*)d_in[25];
  const float* bias_v     = (const float*)d_in[26];
  const float* bias_h     = (const float*)d_in[27];

  const int Euh = in_sizes[1] / 2;   // directed uh edges (already symmetrized)
  const int Ev  = in_sizes[2] / 2;   // video->uh edges
  const int* uh_src = uh_ei;  const int* uh_dst = uh_ei + Euh;
  const int* v_src  = v_ei;   const int* v_dst  = v_ei + Ev;

  // workspace carve
  char* wp = (char*)d_ws;
  auto take = [&](size_t bytes) { char* r = wp; wp += (bytes + 255) & ~(size_t)255; return r; };
  float*    x     = (float*)take((size_t)NN_ * D_ * 4);
  float*    h     = (float*)take((size_t)NN_ * D_ * 4);
  float*    agg   = (float*)take((size_t)NN_ * D_ * 4);
  float*    a_dst = (float*)take((size_t)NN_ * 4);
  float*    a_src = (float*)take((size_t)NN_ * 4);
  float*    ssum  = (float*)take((size_t)NN_ * 4);
  float*    deg   = (float*)take((size_t)NN_ * 4);
  unsigned* m_u   = (unsigned*)take((size_t)NN_ * 4);
  unsigned short* wt_bt = (unsigned short*)take(128 * 512 * 2);
  unsigned short* gatWbt[3], *sageWbt[3];
  for (int i = 0; i < 3; ++i) { gatWbt[i]  = (unsigned short*)take(128 * 128 * 2); }
  for (int i = 0; i < 3; ++i) { sageWbt[i] = (unsigned short*)take(128 * 128 * 2); }
  unsigned short* wv_bt  = (unsigned short*)take(128 * 128 * 2);
  unsigned short* wh_bt  = (unsigned short*)take(128 * 128 * 2);
  unsigned short* wvu_bt = (unsigned short*)take(128 * 128 * 2);
  unsigned short* whu_bt = (unsigned short*)take(128 * 128 * 2);
  float* vidfeat = (float*)take((size_t)B_ * DF_ * 4);
  float* ub   = (float*)take((size_t)B_ * D_ * 4);
  float* posb = (float*)take((size_t)B_ * D_ * 4);
  float* negb = (float*)take((size_t)B_ * D_ * 4);
  float* vid  = (float*)take((size_t)B_ * D_ * 4);
  float* t1   = (float*)take((size_t)B_ * D_ * 4);
  float* t2   = (float*)take((size_t)B_ * D_ * 4);
  float* usv  = (float*)take((size_t)B_ * D_ * 4);
  float* usp  = (float*)take((size_t)B_ * D_ * 4);
  float* usn  = (float*)take((size_t)B_ * D_ * 4);

  const int TPB = 256;
  auto cdiv = [](long a, long b) { return (int)((a + b - 1) / b); };
  auto gemm = [&](const float* A, const unsigned short* Bt, const float* Cadd,
                  const float* bias, float* Out, int rows, int K, int act) {
    int kshift = (K == 512) ? 9 : 7;
    gemm_bf16_wmma<<<(rows + 31) / 32, 256, 0, stream>>>(A, Bt, Cadd, bias, Out, rows, K, kshift, act);
  };

  // weight conversions (bf16, B-transposed layout)
  conv_bf16<<<cdiv(128 * 512, TPB), TPB, 0, stream>>>(W_trans, wt_bt, 128 * 512);
  for (int i = 0; i < 3; ++i) {
    convT128_bf16<<<cdiv(16384, TPB), TPB, 0, stream>>>(gatW[i],  gatWbt[i]);
    convT128_bf16<<<cdiv(16384, TPB), TPB, 0, stream>>>(sageW[i], sageWbt[i]);
  }
  convT128_bf16<<<cdiv(16384, TPB), TPB, 0, stream>>>(weight_v,   wv_bt);
  convT128_bf16<<<cdiv(16384, TPB), TPB, 0, stream>>>(weight_h,   wh_bt);
  convT128_bf16<<<cdiv(16384, TPB), TPB, 0, stream>>>(weight_v_u, wvu_bt);
  convT128_bf16<<<cdiv(16384, TPB), TPB, 0, stream>>>(weight_h_u, whu_bt);

  // x = concat(u_h_embedding, leaky(vf @ W_trans.T + b_trans)); l2norm rows
  copy_f32<<<cdiv((NU_ + NH_) * D_, TPB), TPB, 0, stream>>>(x, uh_emb, (NU_ + NH_) * D_);
  gemm(vf, wt_bt, nullptr, b_trans, x + (size_t)(NU_ + NH_) * D_, NV_, DF_, 1);
  l2norm_rows<<<cdiv((long)NN_ * 32, TPB), TPB, 0, stream>>>(x, NN_);

  for (int l = 0; l < 3; ++l) {
    // ---- GAT on video edges (+ self loops) ----
    gemm(x, gatWbt[l], nullptr, nullptr, h, NN_, D_, 0);
    gat_node_att<<<cdiv((long)NN_ * 32, TPB), TPB, 0, stream>>>(h, gatAtt[l], a_dst, a_src, NN_);
    fill_u32<<<cdiv(NN_, TPB), TPB, 0, stream>>>(m_u, 0x007FFFFFu, NN_);   // encoded(-inf)
    fill_f32<<<cdiv(NN_, TPB), TPB, 0, stream>>>(ssum, 0.f, NN_);
    fill_f32<<<cdiv(NN_ * D_, TPB), TPB, 0, stream>>>(agg, 0.f, NN_ * D_);
    int totv = Ev + NN_;
    gat_alpha_max<<<cdiv(totv, TPB), TPB, 0, stream>>>(v_src, v_dst, Ev, NN_, a_dst, a_src, m_u);
    gat_exp_sum<<<cdiv(totv, TPB), TPB, 0, stream>>>(v_src, v_dst, Ev, NN_, a_dst, a_src, m_u, ssum);
    gat_aggregate<<<cdiv((long)totv * 32, TPB), TPB, 0, stream>>>(v_src, v_dst, Ev, NN_,
                                                                  h, a_dst, a_src, m_u, ssum, agg);
    gat_finalize<<<cdiv(NN_ * D_, TPB), TPB, 0, stream>>>(agg, gatB[l], x, NN_ * D_);

    // ---- SAGE on uh edges (+ self loops) ----
    fill_f32<<<cdiv(NN_, TPB), TPB, 0, stream>>>(deg, 0.f, NN_);
    fill_f32<<<cdiv(NN_ * D_, TPB), TPB, 0, stream>>>(agg, 0.f, NN_ * D_);
    int totu = Euh + NN_;
    sage_aggregate<<<cdiv((long)totu * 32, TPB), TPB, 0, stream>>>(uh_src, uh_dst, Euh, NN_, x, agg, deg);
    sage_norm<<<cdiv(NN_ * D_, TPB), TPB, 0, stream>>>(agg, deg, NN_ * D_);
    gemm(agg, sageWbt[l], nullptr, sageB[l], x, NN_, D_, 1);
  }

  // ---- scoring head ----
  gather_item_rows<<<cdiv(B_ * D_, TPB), TPB, 0, stream>>>(x, item, 0, ub);
  gather_item_rows<<<cdiv(B_ * D_, TPB), TPB, 0, stream>>>(x, item, 2, posb);
  gather_item_rows<<<cdiv(B_ * D_, TPB), TPB, 0, stream>>>(x, item, 3, negb);
  gather_vid<<<cdiv(B_ * DF_, TPB), TPB, 0, stream>>>(vf, item, vidfeat);
  gemm(vidfeat, wt_bt, nullptr, b_trans, vid, B_, DF_, 1);
  gemm(vid, wv_bt,  nullptr, nullptr, t1, B_, D_, 0);
  gemm(ub,  wvu_bt, t1, bias_v, usv, B_, D_, 1);
  gemm(ub,  whu_bt, nullptr, nullptr, t2, B_, D_, 0);
  gemm(posb, wh_bt, t2, bias_h, usp, B_, D_, 1);
  gemm(negb, wh_bt, t2, bias_h, usn, B_, D_, 1);
  scores_kernel<<<cdiv((long)B_ * 32, TPB), TPB, 0, stream>>>(usv, usp, usn, (float*)d_out);
}
